// GraphNetwork_13666585936612
// MI455X (gfx1250) — compile-verified, matrix operand order checked by reading, not run on hardware
//
#include <hip/hip_runtime.h>
#include <hip/hip_bf16.h>

// ---------------------------------------------------------------------------
// Sizes
// ---------------------------------------------------------------------------
static constexpr int BB = 4;      // batch
static constexpr int NN = 96;     // nodes / seq len
static constexpr int FF = 768;    // feature
static constexpr int HH = 384;    // GRU hidden
static constexpr int GG = 3 * HH; // 1152 gates
static constexpr int ROWS = BB * NN;          // 384
static constexpr int K2   = 800;              // F+2=770 padded to mult of 32
static constexpr int KCAT = 2368;             // 3F+POS=2354 padded to mult of 32
static constexpr int POSD = 50;

typedef __attribute__((ext_vector_type(16))) __bf16   v16bf;
typedef __attribute__((ext_vector_type(8)))  float    v8f;
typedef __attribute__((ext_vector_type(8)))  unsigned v8u;

#define DEV __device__ __forceinline__

DEV __bf16 f2bf(float f) {
  unsigned u = __builtin_bit_cast(unsigned, f);
  unsigned r = u + 0x7FFFu + ((u >> 16) & 1u);   // round-to-nearest-even
  unsigned short h = (unsigned short)(r >> 16);
  return __builtin_bit_cast(__bf16, h);
}

// ---------------------------------------------------------------------------
// WMMA bf16 fragment loader (wave32, 16x16x32, per CDNA5 ISA layouts).
// A: 16x32 tile; lane L holds row L%16; reg r holds K pair
//    k0 = 2*(r%4) + (L>=16 ? 8:0) + (r>=4 ? 16:0). Pairs are K-contiguous,
//    so each reg is one dword load. Same mapping serves B when B is given as
//    torch-layout W[N,K] row-major (lane selects output col n = L%16).
// ---------------------------------------------------------------------------
DEV v16bf load_frag(const __bf16* p, int ld) {
  const int lane = threadIdx.x & 31;
  const int row  = lane & 15;
  const int kb   = (lane >> 4) ? 8 : 0;
  v8u u;
#pragma unroll
  for (int r = 0; r < 8; ++r) {
    const int k0 = ((r & 3) << 1) + kb + ((r >> 2) << 4);
    u[r] = *(const unsigned*)(p + (size_t)row * ld + k0);
  }
  return __builtin_bit_cast(v16bf, u);
}

// ---------------------------------------------------------------------------
// Generic GEMM: C[M,N] = A[M,K] @ W[N,K]^T (+bias), A/W bf16, C f32.
// One 16x16 output tile per wave, 8 waves per 256-thread block.
// ---------------------------------------------------------------------------
__global__ __launch_bounds__(256)
void gemm_bf16(const __bf16* __restrict__ A, const __bf16* __restrict__ W,
               float* __restrict__ C, int M, int N, int K,
               const float* __restrict__ bias) {
  const int wave   = blockIdx.x * (blockDim.x >> 5) + ((int)threadIdx.x >> 5);
  const int tilesN = N >> 4;
  const int tiles  = (M >> 4) * tilesN;
  if (wave >= tiles) return;                       // wave-uniform: EXEC all-ones at WMMA
  const int tm = wave / tilesN, tn = wave % tilesN;
  const __bf16* Ab = A + (size_t)tm * 16 * K;
  const __bf16* Wb = W + (size_t)tn * 16 * K;
  v8f acc = {};
  for (int k = 0; k < K; k += 32) {
    v16bf a = load_frag(Ab + k, K);
    v16bf b = load_frag(Wb + k, K);
    acc = __builtin_amdgcn_wmma_f32_16x16x32_bf16(false, a, false, b,
                                                  (short)0, acc, false, false);
  }
  const int lane = threadIdx.x & 31;
  const int col  = tn * 16 + (lane & 15);
  const int row0 = tm * 16 + ((lane >> 4) ? 8 : 0);
  const float bv = bias ? bias[col] : 0.f;
#pragma unroll
  for (int r = 0; r < 8; ++r)
    C[(size_t)(row0 + r) * N + col] = acc[r] + bv;
}

// ---------------------------------------------------------------------------
// f32 -> bf16 with arbitrary row stride / column offset / zero K-padding.
// dst[r*Kp + k] = (k < Ks) ? src[r*ld + off + k] : 0
// ---------------------------------------------------------------------------
__global__ void cvt_gen(const float* __restrict__ src, __bf16* __restrict__ dst,
                        int rows, int Kp, int Ks, int ld, int off) {
  const int idx = blockIdx.x * blockDim.x + threadIdx.x;
  if (idx >= rows * Kp) return;
  const int r = idx / Kp, k = idx % Kp;
  dst[idx] = f2bf(k < Ks ? src[(size_t)r * ld + off + k] : 0.f);
}

// ---------------------------------------------------------------------------
// Persistent bidirectional GRU recurrence. blockIdx.x = direction (0 fwd, 1 bwd).
// h kept in LDS (f32 master + bf16 16-row padded copy for WMMA A fragments).
// gh = h @ whh^T : 72 tiles of 16x16, K=384, across 32 waves, per time step.
// ---------------------------------------------------------------------------
__global__ __launch_bounds__(1024)
void gru_rec(const float* __restrict__ gi_f, const float* __restrict__ gi_b,
             const __bf16* __restrict__ whh_f, const __bf16* __restrict__ whh_b,
             float* __restrict__ y) {
  __shared__ __bf16 hbf[16 * HH];   // 12 KB, rows 4..15 stay zero (M padding)
  __shared__ float  gh [4 * GG];    // 18 KB
  __shared__ float  hf [4 * HH];    //  6 KB
  const int dir  = blockIdx.x;
  const float*  gi  = dir ? gi_b  : gi_f;
  const __bf16* whh = dir ? whh_b : whh_f;
  const int tid  = threadIdx.x;
  const int wave = tid >> 5;
  const int lane = tid & 31;

  for (int p = tid; p < 16 * HH; p += 1024) hbf[p] = f2bf(0.f);
  for (int p = tid; p <  4 * HH; p += 1024) hf[p]  = 0.f;
  __syncthreads();

  for (int t = 0; t < NN; ++t) {
    const int s = dir ? (NN - 1 - t) : t;
    // ---- gh = h @ whh^T (WMMA) ----
    for (int tn = wave; tn < GG / 16; tn += 32) {      // wave-uniform loop
      v8f acc = {};
      for (int kk = 0; kk < HH; kk += 32) {
        v16bf a = load_frag(hbf + kk, HH);
        v16bf b = load_frag(whh + (size_t)tn * 16 * HH + kk, HH);
        acc = __builtin_amdgcn_wmma_f32_16x16x32_bf16(false, a, false, b,
                                                      (short)0, acc, false, false);
      }
      if (lane < 16) {                                  // rows 0..3 valid (batch)
#pragma unroll
        for (int r = 0; r < 4; ++r) gh[r * GG + tn * 16 + lane] = acc[r];
      }
    }
    __syncthreads();
    // ---- gate math + state update ----
    for (int p = tid; p < 4 * HH; p += 1024) {
      const int b = p / HH, j = p % HH;
      const float* g = gi + (size_t)(b * NN + s) * GG;
      const float rr = 1.f / (1.f + __expf(-(g[j]          + gh[b * GG + j])));
      const float zz = 1.f / (1.f + __expf(-(g[HH + j]     + gh[b * GG + HH + j])));
      const float nn = tanhf(g[2 * HH + j] + rr * gh[b * GG + 2 * HH + j]);
      const float h2 = (1.f - zz) * nn + zz * hf[p];
      hf[p] = h2;
      hbf[b * HH + j] = f2bf(h2);
      y[(size_t)(b * NN + s) * FF + dir * HH + j] = h2;
    }
    __syncthreads();
  }
}

// ---------------------------------------------------------------------------
// edge_e / edge_c:  out[row,c] = sum_f sigmoid(lrelu(h[row,f])) * W[c,f] + b[c]
// One wave per row, lane-parallel reduction.
// ---------------------------------------------------------------------------
__global__ __launch_bounds__(256)
void edge_proj(const float* __restrict__ h, const float* __restrict__ W,
               const float* __restrict__ bias, float* __restrict__ out) {
  const int wave = blockIdx.x * (blockDim.x >> 5) + ((int)threadIdx.x >> 5);
  if (wave >= ROWS) return;
  const int lane = threadIdx.x & 31;
  const float* row = h + (size_t)wave * FF;
  float s0 = 0.f, s1 = 0.f;
  for (int f = lane; f < FF; f += 32) {
    float x  = row[f];
    float lr = x >= 0.f ? x : 0.01f * x;
    float sg = 1.f / (1.f + __expf(-lr));
    s0 += sg * W[f];
    s1 += sg * W[FF + f];
  }
#pragma unroll
  for (int o = 16; o; o >>= 1) {
    s0 += __shfl_xor(s0, o, 32);
    s1 += __shfl_xor(s1, o, 32);
  }
  if (lane == 0) {
    out[wave * 2 + 0] = s0 + bias[0];
    out[wave * 2 + 1] = s1 + bias[1];
  }
}

// x2 = [edge_e(2) | node_feat(768) | zeros] -> bf16 [384, 800]
__global__ void build_x2(const float* __restrict__ ee, const float* __restrict__ nf,
                         __bf16* __restrict__ dst) {
  const int idx = blockIdx.x * blockDim.x + threadIdx.x;
  if (idx >= ROWS * K2) return;
  const int r = idx / K2, k = idx % K2;
  float v = 0.f;
  if (k < 2)        v = ee[r * 2 + k];
  else if (k < 770) v = nf[(size_t)r * FF + (k - 2)];
  dst[idx] = f2bf(v);
}

// edge[b,i,j,:] = softmax2(ee[b,i,0]*ec[b,j,0], ee[b,i,1]*ec[b,j,1])
__global__ void init_edge_k(const float* __restrict__ ee, const float* __restrict__ ec,
                            float* __restrict__ edge) {
  const int idx = blockIdx.x * blockDim.x + threadIdx.x;
  if (idx >= BB * NN * NN) return;
  const int b = idx / (NN * NN), rem = idx % (NN * NN), i = rem / NN, j = rem % NN;
  const float e0 = ee[(b * NN + i) * 2 + 0] * ec[(b * NN + j) * 2 + 0];
  const float e1 = ee[(b * NN + i) * 2 + 1] * ec[(b * NN + j) * 2 + 1];
  const float m  = fmaxf(e0, e1);
  const float p0 = __expf(e0 - m), p1 = __expf(e1 - m);
  const float inv = 1.f / (p0 + p1);
  edge[(size_t)idx * 2 + 0] = p0 * inv;
  edge[(size_t)idx * 2 + 1] = p1 * inv;
}

// den[b,j,c] = max(sum_i |edge[b,i,j,c]|, 1e-12)
__global__ void den_k(const float* __restrict__ edge, float* __restrict__ den) {
  const int idx = blockIdx.x * blockDim.x + threadIdx.x;
  if (idx >= BB * NN * 2) return;
  const int b = idx / (NN * 2), rem = idx % (NN * 2), j = rem / 2, c = rem % 2;
  float s = 0.f;
  for (int i = 0; i < NN; ++i)
    s += fabsf(edge[(((size_t)b * NN + i) * NN + j) * 2 + c]);
  den[idx] = fmaxf(s, 1e-12f);
}

// en_bf[b,c,i,j] = bf16( edge[b,i,j,c] / den[b,j,c] )
__global__ void en_k(const float* __restrict__ edge, const float* __restrict__ den,
                     __bf16* __restrict__ en) {
  const int idx = blockIdx.x * blockDim.x + threadIdx.x;
  if (idx >= BB * 2 * NN * NN) return;
  const int b = idx / (2 * NN * NN), rem = idx % (2 * NN * NN);
  const int c = rem / (NN * NN), r2 = rem % (NN * NN), i = r2 / NN, j = r2 % NN;
  en[idx] = f2bf(edge[(((size_t)b * NN + i) * NN + j) * 2 + c] / den[(b * NN + j) * 2 + c]);
}

// nfeT_bf[b,f,j] = bf16(nfe[b,j,f])  (so aggregation GEMM gets K-contiguous B)
__global__ void nfeT_k(const float* __restrict__ nfe, __bf16* __restrict__ dst) {
  const int idx = blockIdx.x * blockDim.x + threadIdx.x;
  if (idx >= BB * FF * NN) return;
  const int b = idx / (FF * NN), rem = idx % (FF * NN), f = rem / NN, j = rem % NN;
  dst[idx] = f2bf(nfe[((size_t)b * NN + j) * FF + f]);
}

// xcat = [nfe | a0 | a1 | pos | zeros] -> bf16 [384, 2368]
__global__ void xcat_k(const float* __restrict__ nfe, const float* __restrict__ a0,
                       const float* __restrict__ a1, const float* __restrict__ pos,
                       __bf16* __restrict__ dst) {
  const int idx = blockIdx.x * blockDim.x + threadIdx.x;
  if (idx >= ROWS * KCAT) return;
  const int r = idx / KCAT, k = idx % KCAT, n = r % NN;
  float v;
  if (k < FF)            v = nfe[(size_t)r * FF + k];
  else if (k < 2 * FF)   v = a0[(size_t)r * FF + (k - FF)];
  else if (k < 3 * FF)   v = a1[(size_t)r * FF + (k - 2 * FF)];
  else if (k < 3 * FF + POSD) v = pos[n * POSD + (k - 3 * FF)];
  else                   v = 0.f;
  dst[idx] = f2bf(v);
}

// LayerNorm over (F,N) per batch + affine(g[f,n], b[f,n]) + lrelu.
__global__ __launch_bounds__(1024)
void layernorm_lrelu(const float* __restrict__ y, const float* __restrict__ g,
                     const float* __restrict__ bta, float* __restrict__ node,
                     __bf16* __restrict__ node_bf) {
  __shared__ float r1[1024], r2[1024];
  const int b = blockIdx.x;
  const float* yb = y + (size_t)b * NN * FF;
  float s = 0.f, s2 = 0.f;
  for (int e = threadIdx.x; e < NN * FF; e += 1024) {
    const float v = yb[e];
    s += v; s2 += v * v;
  }
  r1[threadIdx.x] = s; r2[threadIdx.x] = s2;
  __syncthreads();
  for (int o = 512; o; o >>= 1) {
    if ((int)threadIdx.x < o) {
      r1[threadIdx.x] += r1[threadIdx.x + o];
      r2[threadIdx.x] += r2[threadIdx.x + o];
    }
    __syncthreads();
  }
  const float inv_n = 1.f / (float)(NN * FF);
  const float mu    = r1[0] * inv_n;
  const float var   = r2[0] * inv_n - mu * mu;
  const float rstd  = rsqrtf(var + 1e-5f);
  for (int e = threadIdx.x; e < NN * FF; e += 1024) {
    const int n = e / FF, f = e % FF;
    float v = (yb[e] - mu) * rstd * g[f * NN + n] + bta[f * NN + n];
    v = v >= 0.f ? v : 0.01f * v;
    node[(size_t)b * NN * FF + e]    = v;
    node_bf[(size_t)b * NN * FF + e] = f2bf(v);
  }
}

// Fold (s2 * w2) @ pw^T -> Mf[L,2,768]; cf[L,2] = pw @ b2 + pb
__global__ void mfold_k(const float* __restrict__ pw, const float* __restrict__ bn2g,
                        const float* __restrict__ bn2b, const float* __restrict__ w2,
                        const float* __restrict__ pb, float* __restrict__ Mf,
                        float* __restrict__ cf) {
  const int idx = blockIdx.x * blockDim.x + threadIdx.x;
  if (idx >= 2 * 2 * FF) return;
  const int l = idx / (2 * FF), rem = idx % (2 * FF), c = rem / FF, f1 = rem % FF;
  const float BN = rsqrtf(1.f + 1e-5f);
  float acc = 0.f;
  for (int f2 = 0; f2 < FF; ++f2)
    acc += pw[(l * 2 + c) * FF + f2] * bn2g[l * FF + f2] * BN *
           w2[((size_t)l * FF + f2) * FF + f1];
  Mf[idx] = acc;
  if (f1 == 0) {
    float a = pb[l * 2 + c];
    for (int f2 = 0; f2 < FF; ++f2)
      a += pw[(l * 2 + c) * FF + f2] * bn2b[l * FF + f2];
    cf[l * 2 + c] = a;
  }
}

// Fused rank-structured edge update (replaces both big edge GEMMs):
// h[f] = (A[b,i,f] + B[b,j,f] + e0*w1[f,0] + e1*w1[f,1]) * s1[f] + b1[f]
// out[c] = sum_f lrelu(h[f]) * Mf[c,f] + cf[c]
__global__ __launch_bounds__(128)
void edge_update(const float* __restrict__ edge_in, const float* __restrict__ Amat,
                 const float* __restrict__ Bmat, const float* __restrict__ w1,
                 const float* __restrict__ bn1g, const float* __restrict__ bn1b,
                 const float* __restrict__ Mf, const float* __restrict__ cf,
                 float* __restrict__ edge_out) {
  __shared__ float sA[FF], sW0[FF], sW1[FF], sS[FF], sB[FF], sM0[FF], sM1[FF];
  const int bi = blockIdx.x;           // b*96 + i
  const int b = bi / NN;
  const float BN = rsqrtf(1.f + 1e-5f);
  for (int f = threadIdx.x; f < FF; f += blockDim.x) {
    sA[f]  = Amat[(size_t)bi * FF + f];
    sW0[f] = w1[(size_t)f * 1538 + 0];
    sW1[f] = w1[(size_t)f * 1538 + 1];
    sS[f]  = bn1g[f] * BN;
    sB[f]  = bn1b[f];
    sM0[f] = Mf[f];
    sM1[f] = Mf[FF + f];
  }
  __syncthreads();
  for (int j = threadIdx.x; j < NN; j += blockDim.x) {
    const float* Br = Bmat + ((size_t)b * NN + j) * FF;
    __builtin_prefetch(Br, 0, 1);
    const size_t eidx = ((size_t)bi * NN + j) * 2;
    const float e0 = edge_in[eidx], e1 = edge_in[eidx + 1];
    float a0 = 0.f, a1 = 0.f;
    for (int f = 0; f < FF; ++f) {
      float v = sA[f] + Br[f] + e0 * sW0[f] + e1 * sW1[f];
      v = v * sS[f] + sB[f];
      v = v >= 0.f ? v : 0.01f * v;
      a0 += v * sM0[f];
      a1 += v * sM1[f];
    }
    edge_out[eidx]     = a0 + cf[0];
    edge_out[eidx + 1] = a1 + cf[1];
  }
}

// pre_c / pre_e heads: cat = [nfe | node] (1536), one wave per row.
__global__ __launch_bounds__(256)
void classify(const float* __restrict__ nfe, const float* __restrict__ node,
              const float* __restrict__ cw, const float* __restrict__ cb,
              const float* __restrict__ ew, const float* __restrict__ eb,
              float* __restrict__ out) {
  const int wave = blockIdx.x * (blockDim.x >> 5) + ((int)threadIdx.x >> 5);
  if (wave >= ROWS) return;
  const int lane = threadIdx.x & 31;
  float c0 = 0.f, c1 = 0.f, e0 = 0.f, e1 = 0.f;
  for (int k = lane; k < 2 * FF; k += 32) {
    const float x = (k < FF) ? nfe[(size_t)wave * FF + k]
                             : node[(size_t)wave * FF + (k - FF)];
    c0 += x * cw[k]; c1 += x * cw[2 * FF + k];
    e0 += x * ew[k]; e1 += x * ew[2 * FF + k];
  }
#pragma unroll
  for (int o = 16; o; o >>= 1) {
    c0 += __shfl_xor(c0, o, 32); c1 += __shfl_xor(c1, o, 32);
    e0 += __shfl_xor(e0, o, 32); e1 += __shfl_xor(e1, o, 32);
  }
  if (lane == 0) {
    out[BB * NN * NN * 2 + wave * 2 + 0] = c0 + cb[0];       // pre_c
    out[BB * NN * NN * 2 + wave * 2 + 1] = c1 + cb[1];
    out[BB * NN * NN * 2 + ROWS * 2 + wave * 2 + 0] = e0 + eb[0]; // pre_e
    out[BB * NN * NN * 2 + ROWS * 2 + wave * 2 + 1] = e1 + eb[1];
  }
}

// ---------------------------------------------------------------------------
// Host orchestration
// ---------------------------------------------------------------------------
extern "C" void kernel_launch(void* const* d_in, const int* in_sizes, int n_in,
                              void* d_out, int out_size, void* d_ws, size_t ws_size,
                              hipStream_t stream) {
  (void)in_sizes; (void)n_in; (void)out_size; (void)ws_size;
  const float* node_feat = (const float*)d_in[0];
  const float* g1_wih_f  = (const float*)d_in[2];
  const float* g1_whh_f  = (const float*)d_in[3];
  const float* g1_wih_b  = (const float*)d_in[4];
  const float* g1_whh_b  = (const float*)d_in[5];
  // d_in[6..13] (attention / ref MLPs) are mathematically identity -> unused
  const float* g1_edge_w = (const float*)d_in[14];
  const float* g1_edge_b = (const float*)d_in[15];
  const float* g2_wih_f  = (const float*)d_in[16];
  const float* g2_whh_f  = (const float*)d_in[17];
  const float* g2_wih_b  = (const float*)d_in[18];
  const float* g2_whh_b  = (const float*)d_in[19];
  const float* g2_edge_w = (const float*)d_in[20];
  const float* g2_edge_b = (const float*)d_in[21];
  const float* nd_pos    = (const float*)d_in[22];
  const float* nd_conv_w = (const float*)d_in[23];
  const float* nd_ln_g   = (const float*)d_in[24];
  const float* nd_ln_b   = (const float*)d_in[25];
  const float* ed_w1     = (const float*)d_in[26];
  const float* ed_bn1_g  = (const float*)d_in[27];
  const float* ed_bn1_b  = (const float*)d_in[28];
  const float* ed_w2     = (const float*)d_in[29];
  const float* ed_bn2_g  = (const float*)d_in[30];
  const float* ed_bn2_b  = (const float*)d_in[31];
  const float* ed_pw     = (const float*)d_in[32];
  const float* ed_pb     = (const float*)d_in[33];
  const float* cls_e_w   = (const float*)d_in[34];
  const float* cls_e_b   = (const float*)d_in[35];
  const float* cls_c_w   = (const float*)d_in[36];
  const float* cls_c_b   = (const float*)d_in[37];
  float* out = (float*)d_out;

  char* base = (char*)d_ws;
  size_t off = 0;
  auto alloc = [&](size_t bytes) -> void* {
    void* p = base + off;
    off += (bytes + 255) & ~(size_t)255;
    return p;
  };
  __bf16* Xbf    = (__bf16*)alloc((size_t)ROWS * FF * 2);
  __bf16* wih1f  = (__bf16*)alloc((size_t)GG * FF * 2);
  __bf16* wih1b  = (__bf16*)alloc((size_t)GG * FF * 2);
  __bf16* whh1f  = (__bf16*)alloc((size_t)GG * HH * 2);
  __bf16* whh1b  = (__bf16*)alloc((size_t)GG * HH * 2);
  __bf16* wih2f  = (__bf16*)alloc((size_t)GG * K2 * 2);
  __bf16* wih2b  = (__bf16*)alloc((size_t)GG * K2 * 2);
  __bf16* whh2f  = (__bf16*)alloc((size_t)GG * HH * 2);
  __bf16* whh2b  = (__bf16*)alloc((size_t)GG * HH * 2);
  float*  gi_f   = (float*)alloc((size_t)ROWS * GG * 4);
  float*  gi_b   = (float*)alloc((size_t)ROWS * GG * 4);
  float*  h1     = (float*)alloc((size_t)ROWS * FF * 4);
  float*  nfe    = (float*)alloc((size_t)ROWS * FF * 4);
  float*  edge_e = (float*)alloc((size_t)ROWS * 2 * 4);
  float*  edge_c = (float*)alloc((size_t)ROWS * 2 * 4);
  __bf16* x2bf   = (__bf16*)alloc((size_t)ROWS * K2 * 2);
  float*  edgeA  = (float*)alloc((size_t)BB * NN * NN * 2 * 4);
  float*  edgeB  = (float*)alloc((size_t)BB * NN * NN * 2 * 4);
  float*  den    = (float*)alloc((size_t)BB * NN * 2 * 4);
  __bf16* enbf   = (__bf16*)alloc((size_t)BB * 2 * NN * NN * 2);
  __bf16* nfeTbf = (__bf16*)alloc((size_t)BB * FF * NN * 2);
  float*  a0     = (float*)alloc((size_t)ROWS * FF * 4);
  float*  a1     = (float*)alloc((size_t)ROWS * FF * 4);
  __bf16* xcatbf = (__bf16*)alloc((size_t)ROWS * KCAT * 2);
  __bf16* ndw0   = (__bf16*)alloc((size_t)FF * KCAT * 2);
  __bf16* ndw1   = (__bf16*)alloc((size_t)FF * KCAT * 2);
  float*  ybuf   = (float*)alloc((size_t)ROWS * FF * 4);
  float*  node   = (float*)alloc((size_t)ROWS * FF * 4);
  __bf16* nodebf = (__bf16*)alloc((size_t)ROWS * FF * 2);
  __bf16* W1L    = (__bf16*)alloc((size_t)FF * FF * 2);
  __bf16* W1R    = (__bf16*)alloc((size_t)FF * FF * 2);
  float*  Amat   = (float*)alloc((size_t)ROWS * FF * 4);
  float*  Bmat   = (float*)alloc((size_t)ROWS * FF * 4);
  float*  Mf     = (float*)alloc((size_t)2 * 2 * FF * 4);
  float*  cf     = (float*)alloc((size_t)2 * 2 * 4);

  auto gemm = [&](const __bf16* A, const __bf16* W, float* C, int M, int N, int K,
                  const float* bias) {
    const int tiles  = (M / 16) * (N / 16);
    const int blocks = (tiles + 7) / 8;
    gemm_bf16<<<blocks, 256, 0, stream>>>(A, W, C, M, N, K, bias);
  };
  auto cvt = [&](const float* src, __bf16* dst, int rows, int Kp, int Ks, int ld,
                 int o) {
    const int n = rows * Kp;
    cvt_gen<<<(n + 255) / 256, 256, 0, stream>>>(src, dst, rows, Kp, Ks, ld, o);
  };

  // ---- weight packing (f32 -> bf16, K padded to mult of 32) ----
  cvt(node_feat, Xbf, ROWS, FF, FF, FF, 0);
  cvt(g1_wih_f, wih1f, GG, FF, FF, FF, 0);
  cvt(g1_wih_b, wih1b, GG, FF, FF, FF, 0);
  cvt(g1_whh_f, whh1f, GG, HH, HH, HH, 0);
  cvt(g1_whh_b, whh1b, GG, HH, HH, HH, 0);
  cvt(g2_wih_f, wih2f, GG, K2, 770, 770, 0);
  cvt(g2_wih_b, wih2b, GG, K2, 770, 770, 0);
  cvt(g2_whh_f, whh2f, GG, HH, HH, HH, 0);
  cvt(g2_whh_b, whh2b, GG, HH, HH, HH, 0);
  cvt(nd_conv_w,               ndw0, FF, KCAT, 2354, 2354, 0);
  cvt(nd_conv_w + (size_t)FF * 2354, ndw1, FF, KCAT, 2354, 2354, 0);
  mfold_k<<<(2 * 2 * FF + 255) / 256, 256, 0, stream>>>(ed_pw, ed_bn2_g, ed_bn2_b,
                                                        ed_w2, ed_pb, Mf, cf);

  // ---- GRU1 (attention branches are softmax over size-1 dim == identity) ----
  gemm(Xbf, wih1f, gi_f, ROWS, GG, FF, nullptr);
  gemm(Xbf, wih1b, gi_b, ROWS, GG, FF, nullptr);
  gru_rec<<<2, 1024, 0, stream>>>(gi_f, gi_b, whh1f, whh1b, h1);
  edge_proj<<<ROWS / 8, 256, 0, stream>>>(h1, g1_edge_w, g1_edge_b, edge_e);

  // ---- GRU2 ----
  build_x2<<<(ROWS * K2 + 255) / 256, 256, 0, stream>>>(edge_e, node_feat, x2bf);
  gemm(x2bf, wih2f, gi_f, ROWS, GG, K2, nullptr);
  gemm(x2bf, wih2b, gi_b, ROWS, GG, K2, nullptr);
  gru_rec<<<2, 1024, 0, stream>>>(gi_f, gi_b, whh2f, whh2b, nfe);
  edge_proj<<<ROWS / 8, 256, 0, stream>>>(nfe, g2_edge_w, g2_edge_b, edge_c);

  // ---- init edge + loop-invariant transpose ----
  init_edge_k<<<(BB * NN * NN + 255) / 256, 256, 0, stream>>>(edge_e, edge_c, edgeA);
  nfeT_k<<<(BB * FF * NN + 255) / 256, 256, 0, stream>>>(nfe, nfeTbf);

  // ---- graph layers ----
  for (int l = 0; l < 2; ++l) {
    const float* ein  = l ? edgeB : edgeA;
    float*       eout = l ? out   : edgeB;
    den_k<<<(BB * NN * 2 + 255) / 256, 256, 0, stream>>>(ein, den);
    en_k<<<(BB * 2 * NN * NN + 255) / 256, 256, 0, stream>>>(ein, den, enbf);
    for (int b = 0; b < BB; ++b)
      for (int c = 0; c < 2; ++c)
        gemm(enbf + ((size_t)(b * 2 + c)) * NN * NN,
             nfeTbf + (size_t)b * FF * NN,
             (c ? a1 : a0) + (size_t)b * NN * FF, NN, FF, NN, nullptr);
    xcat_k<<<(ROWS * KCAT + 255) / 256, 256, 0, stream>>>(
        nfe, a0, a1, nd_pos + (size_t)l * NN * POSD, xcatbf);
    gemm(xcatbf, l ? ndw1 : ndw0, ybuf, ROWS, FF, KCAT, nullptr);
    layernorm_lrelu<<<BB, 1024, 0, stream>>>(ybuf, nd_ln_g + (size_t)l * FF * NN,
                                             nd_ln_b + (size_t)l * FF * NN, node,
                                             nodebf);
    cvt(ed_w1 + (size_t)l * FF * 1538, W1L, FF, FF, FF, 1538, 2);
    cvt(ed_w1 + (size_t)l * FF * 1538, W1R, FF, FF, FF, 1538, 770);
    gemm(nodebf, W1L, Amat, ROWS, FF, FF, nullptr);
    gemm(nodebf, W1R, Bmat, ROWS, FF, FF, nullptr);
    edge_update<<<ROWS, 128, 0, stream>>>(ein, Amat, Bmat,
                                          ed_w1 + (size_t)l * FF * 1538,
                                          ed_bn1_g + (size_t)l * FF,
                                          ed_bn1_b + (size_t)l * FF,
                                          Mf + (size_t)l * 2 * FF, cf + l * 2, eout);
  }

  // ---- classification heads ----
  classify<<<ROWS / 8, 256, 0, stream>>>(nfe, node, cls_c_w, cls_c_b, cls_e_w,
                                         cls_e_b, out);
}
